// CrossLayerV2_18811956756866
// MI455X (gfx1250) — matrix operand order repaired and checked by reading, not compiled, not run on hardware
//
#include <hip/hip_runtime.h>
#include <hip/hip_bf16.h>

// ---------------------------------------------------------------------------
// CrossLayerV2 for MI455X (gfx1250, wave32, WMMA, async global->LDS)
//   B=16384, D=512, R=64, E=8
// Pipeline:
//   prep:  V -> f16 (Vh), U -> Uf^T f16 (U2), C -> C^T f16 (Ct)
//   gate:  g = softmax(x @ Wg^T + bg), gsum; also xh = f16(x)
//   gemm1: xvh = f16( xh @ Vh^T )                     [16384x512]x[512x512]
//   gemmC: Yh  = f16( g_e * (xv_e @ C_e) )            block-diag 64x64
//   gemm2: out = x0 * (Yh @ U2^T) + g@b + x*gsum      [16384x512]x[512x512]
// GEMMs: v_wmma_f32_16x16x32_f16, double-buffered LDS fed by
// global_load_async_to_lds_b128 (ASYNCcnt) — CDNA5 async copy path.
// ---------------------------------------------------------------------------

typedef __attribute__((ext_vector_type(16))) _Float16 v16h;
typedef __attribute__((ext_vector_type(8)))  float    v8f;

union FragH { v16h v; uint32_t u[8]; _Float16 h[16]; };

#define Bsz 16384
#define Dd  512
#define Rr  64
#define Ee  8

// CDNA5 async global->LDS copy, 16B per lane. LDS dest address is the low
// 32 bits of the generic pointer (== offset within wave LDS per ISA 10.2).
__device__ __forceinline__ void async_copy_b128(uint32_t lds_off, const void* gaddr) {
    asm volatile("global_load_async_to_lds_b128 %0, %1, off"
                 :: "v"(lds_off), "v"(gaddr)
                 : "memory");
}
__device__ __forceinline__ void wait_async0() {
    asm volatile("s_wait_asynccnt 0" ::: "memory");
}

// ---------------------------------------------------------------------------
// Weight prep: f32 -> f16 with layout changes.
//   Vh[e*64+r][d] = V[e][r][d]   (identity layout, cast only)
//   U2[d][e*64+s] = U[e][d][s]
//   Ct[e][s][r]   = C[e][r][s]
// ---------------------------------------------------------------------------
__global__ __launch_bounds__(256)
void prep_kernel(const float* __restrict__ U, const float* __restrict__ V,
                 const float* __restrict__ C,
                 _Float16* __restrict__ Vh, _Float16* __restrict__ U2,
                 _Float16* __restrict__ Ct) {
    const int i = blockIdx.x * 256 + threadIdx.x;      // [0, 262144)
    Vh[i] = (_Float16)V[i];
    const int e = i >> 15;                             // U: [E][D][R]
    const int d = (i >> 6) & 511;
    const int s = i & 63;
    U2[d * 512 + e * 64 + s] = (_Float16)U[i];
    if (i < Ee * Rr * Rr) {                            // C: [E][R][R]
        const int e2 = i >> 12;
        const int r  = (i >> 6) & 63;
        const int s2 = i & 63;
        Ct[e2 * 4096 + s2 * 64 + r] = (_Float16)C[i];
    }
}

// ---------------------------------------------------------------------------
// Gate: one wave32 per row. logits[e] = dot(x[b,:], Wg[e,:]) + bg[e]; softmax.
// Also emits xh = f16(x) while x is in flight.
// ---------------------------------------------------------------------------
__global__ __launch_bounds__(256)
void gate_kernel(const float* __restrict__ x, const float* __restrict__ Wg,
                 const float* __restrict__ bg, float* __restrict__ g,
                 float* __restrict__ gsum, _Float16* __restrict__ xh) {
    const int wid  = threadIdx.x >> 5;
    const int lane = threadIdx.x & 31;
    const int b    = blockIdx.x * 8 + wid;
    const float* xr = x + (size_t)b * Dd;
    _Float16* xo = xh + (size_t)b * Dd;

    float acc[Ee];
#pragma unroll
    for (int e = 0; e < Ee; ++e) acc[e] = 0.f;

    for (int d = lane; d < Dd; d += 32) {
        const float xv = xr[d];
        xo[d] = (_Float16)xv;
#pragma unroll
        for (int e = 0; e < Ee; ++e)
            acc[e] = fmaf(xv, Wg[e * Dd + d], acc[e]);
    }
#pragma unroll
    for (int e = 0; e < Ee; ++e) {
#pragma unroll
        for (int off = 16; off > 0; off >>= 1)
            acc[e] += __shfl_xor(acc[e], off, 32);
        acc[e] += bg[e];
    }
    float mx = acc[0];
#pragma unroll
    for (int e = 1; e < Ee; ++e) mx = fmaxf(mx, acc[e]);
    float den = 0.f;
#pragma unroll
    for (int e = 0; e < Ee; ++e) { acc[e] = __expf(acc[e] - mx); den += acc[e]; }
    const float inv = 1.f / den;
    float gs = 0.f;
#pragma unroll
    for (int e = 0; e < Ee; ++e) { acc[e] *= inv; gs += acc[e]; }
    if (lane < Ee) g[(size_t)b * Ee + lane] = acc[lane];
    if (lane == 0) gsum[b] = gs;
}

// ---------------------------------------------------------------------------
// Tiled WMMA GEMM: out[row, n0g+col] = sum_k A[row, aCol0+k] * Bt[n0g+col, k]
//   Block tile: 128(M) x 64(N), BK=32, 256 threads = 8 waves (4m x 2n),
//   each wave 32x32 = 2x2 wmma fragments. A and Bt are f16, K-contiguous.
//   Double-buffered LDS, filled by global_load_async_to_lds_b128.
//   EPI 0: outH = f16(acc)                       (gemm1 -> xvh)
//   EPI 1: outH = f16(g[row, blockIdx.y] * acc)  (gemmC -> Yh)
//   EPI 2: outF = x0*acc + sum_e g*b[e] + x*gsum (gemm2 -> out)
// LDS row pitch = 20 dwords (40 halves, 80B): 16B-aligned b128 staging and
// bank-conflict-free (stride 20 mod 64 banks) fragment gathers.
// ---------------------------------------------------------------------------
template<int EPI>
__global__ __launch_bounds__(256)
void gemm_wmma_kernel(const _Float16* __restrict__ A, int lda, int aColPerY,
                      const _Float16* __restrict__ Bt, int ldb, int K,
                      _Float16* __restrict__ outH, float* __restrict__ outF,
                      const float* __restrict__ g, const float* __restrict__ gsum,
                      const float* __restrict__ x0, const float* __restrict__ x,
                      const float* __restrict__ bmat) {
    constexpr int LP = 20;                          // LDS pitch in dwords
    __shared__ __align__(16) uint32_t ldsA[2][128 * LP];
    __shared__ __align__(16) uint32_t ldsB[2][64 * LP];

    const int tid   = threadIdx.x;
    const int row0  = blockIdx.x * 128;
    const int n0g   = blockIdx.y * 64;              // out col base == Bt row base
    const int aCol0 = aColPerY * blockIdx.y;

    const int wid  = tid >> 5;
    const int lane = tid & 31;
    const int wm   = (wid & 3) * 32;                // wave M offset in tile
    const int wn   = (wid >> 2) * 32;               // wave N offset in tile
    const int lg   = lane >> 4;                     // lane group (0/1)
    const int lm   = lane & 15;

    // per-thread staging coordinates (8-half / 16B groups)
    const int mA = tid >> 2;                        // A rows mA and mA+64
    const int kg = tid & 3;                         // 8-half group in k
    const _Float16* aSrc = A + (size_t)row0 * lda + aCol0 + kg * 8;
    const _Float16* bSrc = Bt + (size_t)n0g * ldb + kg * 8;

    const v8f vzero = {0.f, 0.f, 0.f, 0.f, 0.f, 0.f, 0.f, 0.f};
    v8f acc[2][2];
#pragma unroll
    for (int i = 0; i < 2; ++i)
#pragma unroll
        for (int j = 0; j < 2; ++j) acc[i][j] = vzero;

    auto stage = [&](int k0, int buf) {
#pragma unroll
        for (int j = 0; j < 2; ++j) {
            const int m = mA + 64 * j;
            async_copy_b128((uint32_t)(uintptr_t)&ldsA[buf][m * LP + kg * 4],
                            aSrc + (size_t)m * lda + k0);
        }
        async_copy_b128((uint32_t)(uintptr_t)&ldsB[buf][mA * LP + kg * 4],
                        bSrc + (size_t)mA * ldb + k0);
    };

    stage(0, 0);
    wait_async0();
    __syncthreads();

    int buf = 0;
    for (int k0 = 0; k0 < K; k0 += 32) {
        if (k0 + 32 < K) stage(k0 + 32, buf ^ 1);   // prefetch next tile

        // ---- gather fragments per ISA 7.12.2 layouts ----
        FragH a[2], bf[2];
#pragma unroll
        for (int i = 0; i < 2; ++i) {
            const uint32_t* base = &ldsA[buf][(wm + i * 16 + lm) * LP];
#pragma unroll
            for (int v = 0; v < 8; ++v) {
                const int dw = (v < 4) ? (lg * 4 + v) : (8 + lg * 4 + (v - 4));
                a[i].u[v] = base[dw];
            }
        }
#pragma unroll
        for (int j = 0; j < 2; ++j) {
            const uint32_t* base = &ldsB[buf][(wn + j * 16 + lm) * LP];
#pragma unroll
            for (int v = 0; v < 8; ++v)
                bf[j].u[v] = base[(v < 4) ? (lg * 4 + v) : (8 + lg * 4 + (v - 4))];
        }
#pragma unroll
        for (int i = 0; i < 2; ++i)
#pragma unroll
            for (int j = 0; j < 2; ++j)
                acc[i][j] = __builtin_amdgcn_wmma_f32_16x16x32_f16(
                    false, a[i].v, false, bf[j].v, (short)0, acc[i][j],
                    false, false);

        wait_async0();                              // our async writes landed
        __syncthreads();                            // everyone's landed
        buf ^= 1;
    }

    // ---- epilogue: C/D layout: VGPR r -> (M = r + 8*lg, N = lm) ----
#pragma unroll
    for (int j = 0; j < 2; ++j) {
        const int col = n0g + wn + j * 16 + lm;
        float bb[Ee];
        if (EPI == 2) {
#pragma unroll
            for (int e = 0; e < Ee; ++e) bb[e] = bmat[e * Dd + col];
        }
#pragma unroll
        for (int i = 0; i < 2; ++i) {
#pragma unroll
            for (int r = 0; r < 8; ++r) {
                const int row = row0 + wm + i * 16 + lg * 8 + r;
                const float av = acc[i][j][r];
                const size_t idx = (size_t)row * Dd + col;
                if (EPI == 0) {
                    outH[idx] = (_Float16)av;
                } else if (EPI == 1) {
                    outH[idx] = (_Float16)(g[(size_t)row * Ee + blockIdx.y] * av);
                } else {
                    const float* gr = g + (size_t)row * Ee;
                    float s = 0.f;
#pragma unroll
                    for (int e = 0; e < Ee; ++e) s = fmaf(gr[e], bb[e], s);
                    outF[idx] = fmaf(x0[idx], av, s) + x[idx] * gsum[row];
                }
            }
        }
    }
}

// ---------------------------------------------------------------------------
extern "C" void kernel_launch(void* const* d_in, const int* in_sizes, int n_in,
                              void* d_out, int out_size, void* d_ws, size_t ws_size,
                              hipStream_t stream) {
    const float* x0 = (const float*)d_in[0];
    const float* x  = (const float*)d_in[1];
    const float* U  = (const float*)d_in[2];
    const float* V  = (const float*)d_in[3];
    const float* C  = (const float*)d_in[4];
    const float* bm = (const float*)d_in[5];
    const float* Wg = (const float*)d_in[6];
    const float* bg = (const float*)d_in[7];
    float* out = (float*)d_out;

    char* ws = (char*)d_ws;
    size_t o = 0;
    float*    g    = (float*)(ws + o);    o += (size_t)Bsz * Ee * 4;
    float*    gsum = (float*)(ws + o);    o += (size_t)Bsz * 4;
    _Float16* Vh   = (_Float16*)(ws + o); o += (size_t)Dd * Dd * 2;
    _Float16* U2   = (_Float16*)(ws + o); o += (size_t)Dd * Dd * 2;
    _Float16* Ct   = (_Float16*)(ws + o); o += (size_t)Ee * Rr * Rr * 2;
    _Float16* xh   = (_Float16*)(ws + o); o += (size_t)Bsz * Dd * 2;
    _Float16* xvh  = (_Float16*)(ws + o); o += (size_t)Bsz * Dd * 2;
    _Float16* Yh   = (_Float16*)(ws + o); o += (size_t)Bsz * Dd * 2;

    prep_kernel<<<dim3((Ee * Dd * Rr) / 256), 256, 0, stream>>>(U, V, C, Vh, U2, Ct);
    gate_kernel<<<dim3(Bsz / 8), 256, 0, stream>>>(x, Wg, bg, g, gsum, xh);

    // gemm1: xvh = f16(xh @ Vh^T)
    gemm_wmma_kernel<0><<<dim3(Bsz / 128, Dd / 64), 256, 0, stream>>>(
        xh, Dd, 0, Vh, Dd, Dd, xvh, nullptr,
        nullptr, nullptr, nullptr, nullptr, nullptr);

    // gemmC: Yh[b, e*64+s] = f16(g[b,e] * (xv_e @ C_e))
    gemm_wmma_kernel<1><<<dim3(Bsz / 128, Ee), 256, 0, stream>>>(
        xvh, Dd, 64, Ct, Rr, Rr, Yh, nullptr,
        g, nullptr, nullptr, nullptr, nullptr);

    // gemm2 + epilogue: out = x0 * (Yh @ U2^T) + g@b + x*gsum
    gemm_wmma_kernel<2><<<dim3(Bsz / 128, Dd / 64), 256, 0, stream>>>(
        Yh, Dd, 0, U2, Dd, Dd, nullptr, out,
        g, gsum, x0, x, bm);
}
